// WordProbLayer_66211215835769
// MI455X (gfx1250) — matrix operand order repaired and checked by reading, not compiled, not run on hardware
//
#include <hip/hip_runtime.h>
#include <hip/hip_bf16.h>
#include <math.h>

#define T_LEN 64
#define BATCH 32
#define SEQ   400
#define HDIM  512
#define VOCAB 50000
#define EXTL  400
#define CDIM  (3*HDIM)

typedef __attribute__((ext_vector_type(16))) __bf16 v16bf;
typedef __attribute__((ext_vector_type(8)))  float  v8f;

// two floats -> packed bf16 pair (round-half-up) in 3 VALU ops via v_perm_b32
__device__ __forceinline__ unsigned int bfpack2(float x, float y) {
  union { float f; unsigned int u; } a, b; a.f = x; b.f = y;
  unsigned int xu = a.u + 0x8000u;
  unsigned int yu = b.u + 0x8000u;
  return __builtin_amdgcn_perm(yu, xu, 0x07060302u);
}

__device__ __forceinline__ uint2 pack4(float4 v4) {
  uint2 r;
  r.x = bfpack2(v4.x, v4.y);
  r.y = bfpack2(v4.z, v4.w);
  return r;
}

// A-frag (16x32 bf16): lane M = lane&15; lanes0-15 K=0-7/16-23, lanes16-31 +8.
__device__ __forceinline__ v16bf load_a_frag(const unsigned short* __restrict__ As,
                                             int mbase, int lane) {
  const int row  = mbase + (lane & 15);
  const int half = lane >> 4;
  union { v16bf v; uint4 q[2]; } r;
  const unsigned short* p = As + row * 32 + half * 8;
  r.q[0] = *(const uint4*)(p);        // K = half*8 .. half*8+7
  r.q[1] = *(const uint4*)(p + 16);   // K = 16+half*8 .. +7
  return r.v;
}

// B-frag (32x16 bf16): lane N = lane&15; lanes0-15 K=0-7/16-23, lanes16-31 +8.
__device__ __forceinline__ v16bf load_b_frag(const unsigned short* __restrict__ Bs,
                                             int K, int nbase, int kb, int lane) {
  const int col  = nbase + (lane & 15);
  const int half = lane >> 4;
  union { v16bf v; uint4 q[2]; } r;
  const unsigned short* p = Bs + col * K + kb + half * 8;
  r.q[0] = *(const uint4*)(p);
  r.q[1] = *(const uint4*)(p + 16);
  return r.v;
}

// A chunk 256x32 f32 -> bf16 LDS; split load/store phases so 8 global_load_b128
// pipeline before any wait.
__device__ __forceinline__ void fill_a_chunk(unsigned short* __restrict__ dst,
                                             const float* __restrict__ A,
                                             int m0, int K, int kb, int tid) {
  float4 t[8];
#pragma unroll
  for (int i = 0; i < 8; ++i) {
    int off = (tid + i * 256) << 2;             // 0..8191
    int r = off >> 5, cc = off & 31;
    t[i] = *(const float4*)(A + (size_t)(m0 + r) * K + kb + cc);
  }
#pragma unroll
  for (int i = 0; i < 8; ++i) {
    int off = (tid + i * 256) << 2;
    int r = off >> 5, cc = off & 31;
    *(uint2*)(dst + r * 32 + cc) = pack4(t[i]);
  }
}

// C[M,N] = (A[M,K] @ B[N,K]^T + bias[N]) * alpha ; A,B f32 -> bf16 on LDS fill.
// 256 thr = 8 waves; BM=256, BN=64; wave w owns rows w*32..+31, all 64 cols
// (2 A-frags x 4 B-frags = 8 WMMA / K-step). Full-K B tile resident in LDS;
// A chunk double-buffered (1 barrier per K-step); block persists over M tiles.
// Requires M % 256 == 0, K % 32 == 0. N guarded.
__global__ __launch_bounds__(256)
void gemm_bf16_wmma(const float* __restrict__ A, const float* __restrict__ B,
                    const float* __restrict__ bias, float* __restrict__ C,
                    int M, int N, int K, long long ldc, float alpha) {
  extern __shared__ unsigned short smem[];
  unsigned short* As0 = smem;            // 256*32 bf16 (16 KB)
  unsigned short* As1 = smem + 8192;     // 256*32 bf16 (16 KB)
  unsigned short* Bs  = smem + 16384;    // 64*K bf16

  const int tid  = threadIdx.x;
  const int wid  = tid >> 5;
  const int lane = tid & 31;
  const int wm   = wid * 32;
  const int n0   = blockIdx.x * 64;
  const int half = lane >> 4;

  // ---- full-K B tile (bf16), once per block; batched split-phase fill ----
  const int iters = (64 * K) >> 10;             // float4s per thread
  for (int it = 0; it < iters; it += 4) {
    float4 t[4];
#pragma unroll
    for (int u = 0; u < 4; ++u) {
      int off = ((it + u) * 256 + tid) << 2;
      int j = off / K, kk = off % K;
      t[u] = make_float4(0.f, 0.f, 0.f, 0.f);
      if (n0 + j < N) t[u] = *(const float4*)(B + (size_t)(n0 + j) * K + kk);
    }
#pragma unroll
    for (int u = 0; u < 4; ++u) {
      int off = ((it + u) * 256 + tid) << 2;
      int j = off / K, kk = off % K;
      *(uint2*)(Bs + j * K + kk) = pack4(t[u]);
    }
  }

  // hoist column state (invariant over M tiles)
  int   ncol[4]; bool nok[4]; float bvv[4];
#pragma unroll
  for (int bi = 0; bi < 4; ++bi) {
    ncol[bi] = n0 + bi * 16 + (lane & 15);
    nok[bi]  = ncol[bi] < N;
    bvv[bi]  = (nok[bi] && bias) ? bias[ncol[bi]] : 0.f;
  }

  const int MT = M >> 8;
  for (int mt = blockIdx.y; mt < MT; mt += gridDim.y) {
    const int m0 = mt << 8;
    v8f acc[2][4];
#pragma unroll
    for (int i = 0; i < 2; ++i)
#pragma unroll
      for (int j = 0; j < 4; ++j) acc[i][j] = (v8f){0.f,0.f,0.f,0.f,0.f,0.f,0.f,0.f};

    __syncthreads();                       // prior reads (B fill / prev mt) done
    fill_a_chunk(As0, A, m0, K, 0, tid);   // prologue: chunk 0

    for (int kb = 0; kb < K; kb += 32) {
      unsigned short* cur = (kb & 32) ? As1 : As0;
      unsigned short* nxt = (kb & 32) ? As0 : As1;
      __syncthreads();                     // cur chunk visible to all waves
      if (kb + 32 < K)                     // overlap next-chunk fill with WMMAs
        fill_a_chunk(nxt, A, m0, K, kb + 32, tid);

      v16bf bfr[4];
#pragma unroll
      for (int bi = 0; bi < 4; ++bi) bfr[bi] = load_b_frag(Bs, K, bi * 16, kb, lane);
#pragma unroll
      for (int ai = 0; ai < 2; ++ai) {
        v16bf af = load_a_frag(cur, wm + ai * 16, lane);
#pragma unroll
        for (int bi = 0; bi < 4; ++bi)
          acc[ai][bi] = __builtin_amdgcn_wmma_f32_16x16x32_bf16(
              false, af, false, bfr[bi], (short)0, acc[ai][bi], false, false);
      }
    }

#pragma unroll
    for (int ai = 0; ai < 2; ++ai) {
      const int mrow0 = m0 + wm + ai * 16 + half * 8;
#pragma unroll
      for (int bi = 0; bi < 4; ++bi) {
        if (nok[bi]) {
          float* cp = C + (size_t)mrow0 * ldc + ncol[bi];
#pragma unroll
          for (int vj = 0; vj < 8; ++vj)
            cp[(size_t)vj * ldc] = (acc[ai][bi][vj] + bvv[bi]) * alpha;
        }
      }
    }
  }
}

// one block per (t,b): scores -> softmax -> dists + weighted sum of V-proj
__global__ __launch_bounds__(256)
void attn_kernel(const float* __restrict__ q, const float* __restrict__ k,
                 const float* __restrict__ vproj, const unsigned char* __restrict__ mask,
                 float* __restrict__ attsum, float* __restrict__ dists) {
  const int m = blockIdx.x;           // m = t*B + b
  const int b = m % BATCH;
  const int tid = threadIdx.x;
  __shared__ float sw[SEQ];
  __shared__ float red[256];

  const float* qr = q + (size_t)m * HDIM;
  for (int s = tid; s < SEQ; s += 256) {
    const float* kr = k + (size_t)(s * BATCH + b) * HDIM;
    float a = 0.f;
    for (int hh = 0; hh < HDIM; hh += 4) {
      float4 qa = *(const float4*)(qr + hh);
      float4 ka = *(const float4*)(kr + hh);
      a += qa.x * ka.x + qa.y * ka.y + qa.z * ka.z + qa.w * ka.w;
    }
    if (mask[b * SEQ + s]) a = -1e9f;
    sw[s] = a;
  }
  __syncthreads();

  float mx = -1e30f;
  for (int s = tid; s < SEQ; s += 256) mx = fmaxf(mx, sw[s]);
  red[tid] = mx; __syncthreads();
  for (int st = 128; st > 0; st >>= 1) {
    if (tid < st) red[tid] = fmaxf(red[tid], red[tid + st]);
    __syncthreads();
  }
  const float rowmax = red[0];
  __syncthreads();

  float sum = 0.f;
  for (int s = tid; s < SEQ; s += 256) {
    float e = __expf(sw[s] - rowmax);
    sw[s] = e; sum += e;
  }
  red[tid] = sum; __syncthreads();
  for (int st = 128; st > 0; st >>= 1) {
    if (tid < st) red[tid] += red[tid + st];
    __syncthreads();
  }
  const float inv = 1.f / red[0];
  __syncthreads();

  for (int s = tid; s < SEQ; s += 256) {
    float w = sw[s] * inv;
    sw[s] = w;
    dists[(size_t)m * SEQ + s] = w;
  }
  __syncthreads();

  for (int hh = tid; hh < HDIM; hh += 256) {
    float a = 0.f;
    for (int s = 0; s < SEQ; ++s)
      a += sw[s] * vproj[(size_t)(s * BATCH + b) * HDIM + hh];
    attsum[(size_t)m * HDIM + hh] = a;
  }
}

// cat[:, 0:512]=h, cat[:, 512:1024]=y_emb (atts slice written by Wo GEMM)
__global__ __launch_bounds__(256)
void cat_copy(const float* __restrict__ h, const float* __restrict__ y,
              float* __restrict__ cat) {
  long long i = (long long)blockIdx.x * 256 + threadIdx.x;
  const long long total = (long long)T_LEN * BATCH * 2 * HDIM;
  if (i >= total) return;
  int m = (int)(i >> 10);
  int c = (int)(i & 1023);
  float v = (c < HDIM) ? h[(size_t)m * HDIM + c] : y[(size_t)m * HDIM + (c - HDIM)];
  cat[(size_t)m * CDIM + c] = v;
}

__global__ __launch_bounds__(256)
void gate_kernel(const float* __restrict__ cat, const float* __restrict__ vvec,
                 const float* __restrict__ bv, float* __restrict__ g) {
  const int m = blockIdx.x;
  const int tid = threadIdx.x;
  __shared__ float red[256];
  float a = 0.f;
  for (int c = tid; c < CDIM; c += 256) a += cat[(size_t)m * CDIM + c] * vvec[c];
  red[tid] = a; __syncthreads();
  for (int st = 128; st > 0; st >>= 1) {
    if (tid < st) red[tid] += red[tid + st];
    __syncthreads();
  }
  if (tid == 0) g[m] = 1.f / (1.f + __expf(-(red[0] + bv[0])));
}

// per row: stage logits in 320KB LDS, online softmax over V (single global read),
// rescale*g from LDS, zero EXT tail, scatter (1-g)*dists
__global__ __launch_bounds__(256)
void softmax_scatter(float* __restrict__ pred, const float* __restrict__ dists,
                     const float* __restrict__ g, const int* __restrict__ xids) {
  extern __shared__ float dsm[];           // VOCAB floats + 512 reduce scratch
  float* rowl = dsm;                       // 50000
  float* rm   = dsm + VOCAB;               // 256
  float* rl   = rm + 256;                  // 256

  const int m = blockIdx.x;
  const int b = m % BATCH;
  const int tid = threadIdx.x;
  float* row = pred + (size_t)m * (VOCAB + EXTL);

  float mi = -1e30f, li = 0.f;
  for (int n = tid; n < VOCAB; n += 256) {
    float x = row[n];
    rowl[n] = x;
    if (x > mi) { li *= __expf(mi - x); mi = x; }
    li += __expf(x - mi);
  }
  rm[tid] = mi; rl[tid] = li; __syncthreads();
  for (int st = 128; st > 0; st >>= 1) {
    if (tid < st) {
      float m2 = rm[tid + st], l2 = rl[tid + st];
      float mo = fmaxf(rm[tid], m2);
      rl[tid] = rl[tid] * __expf(rm[tid] - mo) + l2 * __expf(m2 - mo);
      rm[tid] = mo;
    }
    __syncthreads();
  }
  const float rmax = rm[0];
  const float gm = g[m];
  const float scale = gm / rl[0];

  for (int n = tid; n < VOCAB; n += 256) row[n] = __expf(rowl[n] - rmax) * scale;
  for (int n = VOCAB + tid; n < VOCAB + EXTL; n += 256) row[n] = 0.f;
  __syncthreads();

  const float og = 1.f - gm;
  for (int s = tid; s < SEQ; s += 256) {
    int idx = xids[s * BATCH + b];
    atomicAdd(row + idx, og * dists[(size_t)m * SEQ + s]);
  }
}

extern "C" void kernel_launch(void* const* d_in, const int* in_sizes, int n_in,
                              void* d_out, int out_size, void* d_ws, size_t ws_size,
                              hipStream_t stream) {
  (void)in_sizes; (void)n_in; (void)out_size; (void)ws_size;
  const float* h      = (const float*)d_in[0];
  const float* y_emb  = (const float*)d_in[1];
  const float* memory = (const float*)d_in[2];
  const unsigned char* mask = (const unsigned char*)d_in[3];
  const int*   xids   = (const int*)d_in[4];
  const float* Wq = (const float*)d_in[6],  *bq = (const float*)d_in[7];
  const float* Wk = (const float*)d_in[8],  *bk = (const float*)d_in[9];
  const float* Wv = (const float*)d_in[10], *bva = (const float*)d_in[11];
  const float* Wo = (const float*)d_in[12], *bo = (const float*)d_in[13];
  const float* Wp = (const float*)d_in[14], *bp = (const float*)d_in[15];
  const float* vvec = (const float*)d_in[16], *bvg = (const float*)d_in[17];

  const int M  = T_LEN * BATCH;        // 2048
  const int MS = SEQ * BATCH;          // 12800

  float* qws    = (float*)d_ws;
  float* kws    = qws    + (size_t)M  * HDIM;
  float* vws    = kws    + (size_t)MS * HDIM;
  float* attsum = vws    + (size_t)MS * HDIM;
  float* catws  = attsum + (size_t)M  * HDIM;
  float* gws    = catws  + (size_t)M  * CDIM;

  float* pred  = (float*)d_out;
  float* dists = pred + (size_t)M * (VOCAB + EXTL);

  const size_t shProj = (size_t)(2 * 256 * 32 + 64 * HDIM) * 2;  //  98,304 B
  const size_t shBig  = (size_t)(2 * 256 * 32 + 64 * CDIM) * 2;  // 229,376 B
  const size_t shSm   = (size_t)(VOCAB + 512) * 4;               // 202,048 B
  const float qscale = 0.04419417382415922f;                     // 1/sqrt(512)

  // QKV projections (bf16 WMMA); M tiles of 256
  gemm_bf16_wmma<<<dim3(HDIM / 64, 8), 256, shProj, stream>>>(
      h, Wq, bq, qws, M, HDIM, HDIM, HDIM, qscale);
  gemm_bf16_wmma<<<dim3(HDIM / 64, 50), 256, shProj, stream>>>(
      memory, Wk, bk, kws, MS, HDIM, HDIM, HDIM, 1.0f);
  gemm_bf16_wmma<<<dim3(HDIM / 64, 50), 256, shProj, stream>>>(
      memory, Wv, bva, vws, MS, HDIM, HDIM, HDIM, 1.0f);

  // cat[:,0:1024] = [h, y_emb]
  {
    long long total = (long long)M * 2 * HDIM;
    cat_copy<<<dim3((unsigned)((total + 255) / 256)), 256, 0, stream>>>(h, y_emb, catws);
  }

  // attention: scores/softmax/dists + weighted V sum
  attn_kernel<<<dim3(M), 256, 0, stream>>>(qws, kws, vws, mask, attsum, dists);

  // Wo projection, written straight into cat[:,1024:1536] via ldc=1536
  gemm_bf16_wmma<<<dim3(HDIM / 64, 8), 256, shProj, stream>>>(
      attsum, Wo, bo, catws + 2 * HDIM, M, HDIM, HDIM, CDIM, 1.0f);

  // pointer-generator gate
  gate_kernel<<<dim3(M), 256, 0, stream>>>(catws, vvec, bvg, gws);

  // big projection: logits into d_out (ldc = V+EXT); Wp read only gridDim.y=2 times
  gemm_bf16_wmma<<<dim3((VOCAB + 63) / 64, 2), 256, shBig, stream>>>(
      catws, Wp, bp, pred, M, VOCAB, CDIM, VOCAB + EXTL, 1.0f);

  // softmax over V (LDS-staged, single global read), scale by g, zero EXT, scatter
  softmax_scatter<<<dim3(M), 256, shSm, stream>>>(pred, dists, gws, xids);
}
